// LSHSelfAttention_29051158790686
// MI455X (gfx1250) — matrix operand order repaired and checked by reading, not compiled, not run on hardware
//
#include <hip/hip_runtime.h>
#include <hip/hip_bf16.h>

// Problem constants
#define BB     2
#define SS     4096
#define HIDN   1024
#define HH     16
#define DD     64
#define NHASH  4
#define NBUCK  64            // buckets per hash
#define TBUCK  (NBUCK*NHASH) // 256 distinct bucket ids
#define LCH    64            // chunk
#define LL     (NHASH*SS)    // 16384
#define NCHUNK (LL/LCH)      // 256
#define BH     (BB*HH)       // 32

typedef __attribute__((ext_vector_type(16))) _Float16 v16h;
typedef __attribute__((ext_vector_type(8)))  float    v8f;

union FragA { v16h v; unsigned u[8]; };

static __device__ __forceinline__ unsigned pack2(float a, float b) {
  union { unsigned u; _Float16 h[2]; } p;
  p.h[0] = (_Float16)a; p.h[1] = (_Float16)b;
  return p.u;
}

// ---------------------------------------------------------------------------
// K0: zero the histograms
// ---------------------------------------------------------------------------
__global__ void k_zero(int* __restrict__ p, int n) {
  int i = blockIdx.x * 256 + threadIdx.x;
  if (i < n) p[i] = 0;
}

// ---------------------------------------------------------------------------
// K1: fused QK & V projection.  C = hidden(8192x1024) x W(1024x1024), two Ws
// sharing the A tile.  Output layout (B,H,S,D) f32.
// block = 256 thr (8 waves), tile 64(M) x 64(N), K-step 32.
// Staging loops fully unrolled so all global loads are in flight before any
// wait; global_prefetch_b8 pulls the next K-step tiles under the WMMA loop.
// ---------------------------------------------------------------------------
__global__ __launch_bounds__(256)
void k_proj(const float* __restrict__ hid, const float* __restrict__ wqk,
            const float* __restrict__ wv, float* __restrict__ qk,
            float* __restrict__ vout)
{
  __shared__ unsigned As[64][16];   // [row][kpair] packed f16
  __shared__ unsigned Bq[64][16];   // [n][kpair]  (W transposed)
  __shared__ unsigned Bv[64][16];

  const int tid  = threadIdx.x;
  const int lane = tid & 31;
  const int w    = tid >> 5;     // 0..7
  const int wn   = w & 3;        // N subtile (16 cols each)
  const int wm   = w >> 2;       // 0..1 -> rows 32*wm
  const int li   = lane & 15;
  const int hi   = lane >> 4;
  const int rowBase = blockIdx.x * 64;
  const int colBase = blockIdx.y * 64;

  v8f accq0 = {0,0,0,0,0,0,0,0}, accq1 = {0,0,0,0,0,0,0,0};
  v8f accv0 = {0,0,0,0,0,0,0,0}, accv1 = {0,0,0,0,0,0,0,0};

  for (int k0 = 0; k0 < HIDN; k0 += 32) {
    // stage A (64 rows x 32 K) as f16 pairs -- 4 iters, fully unrolled
#pragma unroll
    for (int it = 0; it < 4; ++it) {
      int p = tid + it * 256;
      int r = p >> 4, kp = p & 15;
      const float* src = hid + (size_t)(rowBase + r) * HIDN + k0 + 2 * kp;
      As[r][kp] = pack2(src[0], src[1]);
    }
    // stage both Ws transposed: Bx[n][kpair] -- 4 iters, fully unrolled
#pragma unroll
    for (int it = 0; it < 4; ++it) {
      int p = tid + it * 256;
      int n = p >> 4, kp = p & 15;
      size_t o = (size_t)(k0 + 2 * kp) * HIDN + colBase + n;
      Bq[n][kp] = pack2(wqk[o], wqk[o + HIDN]);
      Bv[n][kp] = pack2(wv[o],  wv[o + HIDN]);
    }
    // prefetch next K-step tiles (global_prefetch_b8; cacheline granular)
    int nk = k0 + 32;
    if (nk < HIDN) {
      __builtin_prefetch(hid + (size_t)(rowBase + (tid & 63)) * HIDN + nk, 0, 3);
      __builtin_prefetch(wqk + (size_t)(nk + (tid & 31)) * HIDN + colBase, 0, 3);
      __builtin_prefetch(wv  + (size_t)(nk + (tid & 31)) * HIDN + colBase, 0, 3);
    }
    __syncthreads();

    FragA a0, a1, bq, bv;
#pragma unroll
    for (int vv = 0; vv < 8; ++vv) {
      int kpA = (vv & 3) + 4 * hi + 8 * (vv >> 2);
      a0.u[vv] = As[32 * wm + li][kpA];
      a1.u[vv] = As[32 * wm + 16 + li][kpA];
      int kpB = (vv & 3) + 4 * hi + 8 * (vv >> 2);
      bq.u[vv] = Bq[16 * wn + li][kpB];
      bv.u[vv] = Bv[16 * wn + li][kpB];
    }
    accq0 = __builtin_amdgcn_wmma_f32_16x16x32_f16(false, a0.v, false, bq.v, (short)0, accq0, false, false);
    accq1 = __builtin_amdgcn_wmma_f32_16x16x32_f16(false, a1.v, false, bq.v, (short)0, accq1, false, false);
    accv0 = __builtin_amdgcn_wmma_f32_16x16x32_f16(false, a0.v, false, bv.v, (short)0, accv0, false, false);
    accv1 = __builtin_amdgcn_wmma_f32_16x16x32_f16(false, a1.v, false, bv.v, (short)0, accv1, false, false);
    __syncthreads();
  }

  // store to (B,H,S,D)
#pragma unroll
  for (int t2 = 0; t2 < 2; ++t2) {
#pragma unroll
    for (int vv = 0; vv < 8; ++vv) {
      int gr = rowBase + 32 * wm + 16 * t2 + vv + 8 * hi; // global row in [0,8192)
      int b  = gr >> 12;
      int s  = gr & (SS - 1);
      int c  = colBase + 16 * wn + li;
      int h  = c >> 6, d = c & 63;
      size_t o = ((size_t)(b * HH + h) * SS + s) * DD + d;
      qk[o]   = t2 ? accq1[vv] : accq0[vv];
      vout[o] = t2 ? accv1[vv] : accv0[vv];
    }
  }
}

// ---------------------------------------------------------------------------
// K2: LSH bucketing. rot[b,h,n,s,r] = sum_d qk*rotations; argmax over
// [rot,-rot] (first occurrence ties) -> bucket + 64n.  Also histogram.
// One thread per (b,h,s). f32 VALU (precision-sensitive argmax).
// ---------------------------------------------------------------------------
__global__ __launch_bounds__(256)
void k_buckets(const float* __restrict__ qk, const float* __restrict__ rotations,
               int* __restrict__ buckets, int* __restrict__ hist)
{
  __shared__ float rot[DD * NHASH * 32]; // 32 KB
  for (int i = threadIdx.x; i < DD * NHASH * 32; i += 256) rot[i] = rotations[i];
  __syncthreads();

  int gid = blockIdx.x * 256 + threadIdx.x;  // bh*S + s
  int bh  = gid >> 12;
  int s   = gid & (SS - 1);

  float q[DD];
  const float* qr = qk + (size_t)gid * DD;
#pragma unroll
  for (int d = 0; d < DD; ++d) q[d] = qr[d];

  for (int n = 0; n < NHASH; ++n) {
    float mx = -1e30f, mn = 1e30f;
    int amx = 0, amn = 0;
    for (int r = 0; r < 32; ++r) {
      float acc = 0.f;
#pragma unroll
      for (int d = 0; d < DD; ++d) acc = fmaf(q[d], rot[d * 128 + n * 32 + r], acc);
      if (acc > mx) { mx = acc; amx = r; }
      if (acc < mn) { mn = acc; amn = r; }
    }
    int idx   = (mx >= -mn) ? amx : (32 + amn); // argmax over concat([rot,-rot])
    int bfull = idx + NBUCK * n;
    buckets[(size_t)bh * LL + n * SS + s] = bfull;
    atomicAdd(&hist[bh * TBUCK + bfull], 1);
  }
}

// ---------------------------------------------------------------------------
// K3: exact stable counting sort per (b,h): prefix over 256 bins, then
// bucket-owner stable scatter (thread j owns bucket j).  Emits sorted tickers.
// ---------------------------------------------------------------------------
__global__ __launch_bounds__(256)
void k_sort(const int* __restrict__ buckets, const int* __restrict__ hist,
            int* __restrict__ sorted_t)
{
  __shared__ unsigned short lb[LL];   // 32 KB
  __shared__ int pre[TBUCK + 1];
  const int bh = blockIdx.x;

  for (int i = threadIdx.x; i < LL; i += 256)
    lb[i] = (unsigned short)buckets[(size_t)bh * LL + i];
  if (threadIdx.x == 0) {
    int acc = 0;
    for (int j = 0; j < TBUCK; ++j) { pre[j] = acc; acc += hist[bh * TBUCK + j]; }
    pre[TBUCK] = acc;
  }
  __syncthreads();

  const int j = threadIdx.x;     // owned bucket
  int off = pre[j];
  for (int t = 0; t < LL; ++t) {
    if (lb[t] == (unsigned short)j) { sorted_t[(size_t)bh * LL + off] = t; ++off; }
  }
}

// ---------------------------------------------------------------------------
// K4: chunked attention.  Per (chunk, bh): async-gather tickers into LDS,
// gather Q(64xD) raw and K(128xD) RMS-normalized, V(128xD).
// dots = Q*K^T via WMMA, masks, in-wave logsumexp, probs*V via WMMA,
// scatter out/logits by (hash, pos) (fuses undo-sort).
// block = 128 thr (4 waves), one 16-row M-tile per wave.
// ---------------------------------------------------------------------------
__global__ __launch_bounds__(128)
void k_attn(const float* __restrict__ qk, const float* __restrict__ v,
            const int* __restrict__ sorted_t, float* __restrict__ outh,
            float* __restrict__ logitsg)
{
  __shared__ int tk[192];             // [0:128) = K window tickers, [64:128) dup'd
                                      // as Q tickers via tq = tk+64
  __shared__ unsigned Qu[64][32];     // q rows, f16 pairs over d   (8 KB)
  __shared__ unsigned Ku[128][32];    // key rows (normalized)      (16 KB)
  __shared__ _Float16 Vh[64][128];    // V^T: [d][j]                (16 KB)
  __shared__ _Float16 Ph[64][128];    // probs: [q][j]              (16 KB)

  const int tid   = threadIdx.x;
  const int c     = blockIdx.x;
  const int bh    = blockIdx.y;
  const int cprev = (c + NCHUNK - 1) & (NCHUNK - 1);
  int* tq = tk + 64;                  // current chunk tickers (q rows)

  // Per-lane async gather of the 128-wide ticker window straight into LDS
  // (GLOBAL_LOAD_ASYNC_TO_LDS_B32, tracked by ASYNCcnt).
  {
    int idx = (tid < 64) ? (cprev * 64 + tid) : (c * 64 + (tid - 64));
    const int* gsrc = sorted_t + (size_t)bh * LL + idx;
    unsigned lds_off = (unsigned)(size_t)&tk[tid];   // low 32 bits = LDS byte offset
    asm volatile("global_load_async_to_lds_b32 %0, %1, off"
                 :: "v"(lds_off), "v"((unsigned long long)(size_t)gsrc)
                 : "memory");
    asm volatile("s_wait_asynccnt 0x0" ::: "memory");
  }
  __syncthreads();

  // Q rows (raw qk), packed f16 -- fully unrolled so loads overlap
#pragma unroll
  for (int it = 0; it < 16; ++it) {
    int p = tid + it * 128;
    int r = p >> 5, kp = p & 31;
    const float* src = qk + ((size_t)bh * SS + (tq[r] & (SS - 1))) * DD + 2 * kp;
    Qu[r][kp] = pack2(src[0], src[1]);
  }
  // K rows normalized (rsqrt(mean sq + 1e-6)/sqrt(D)) + V rows transposed
  {
    int j   = tid;
    int pos = tk[j] & (SS - 1);
    const float* kr = qk + ((size_t)bh * SS + pos) * DD;
    float rowv[DD];
    float ssum = 0.f;
#pragma unroll
    for (int d = 0; d < DD; ++d) { rowv[d] = kr[d]; ssum = fmaf(rowv[d], rowv[d], ssum); }
    float scale = rsqrtf(ssum * (1.0f / DD) + 1e-6f) * 0.125f;
#pragma unroll
    for (int kp = 0; kp < 32; ++kp)
      Ku[j][kp] = pack2(rowv[2 * kp] * scale, rowv[2 * kp + 1] * scale);
    const float* vr = v + ((size_t)bh * SS + pos) * DD;
#pragma unroll
    for (int d = 0; d < DD; ++d) Vh[d][j] = (_Float16)vr[d];
  }
  __syncthreads();

  const int w    = tid >> 5;      // wave: M-tile rows [16w,16w+16)
  const int lane = tid & 31;
  const int li   = lane & 15;
  const int hi   = lane >> 4;

  // dots = Q(64x64) x K^T -> 64x128, K-dim = D = 64 (2 wmma k-steps)
  v8f acc[8];
#pragma unroll
  for (int nt = 0; nt < 8; ++nt) acc[nt] = (v8f){0,0,0,0,0,0,0,0};
#pragma unroll
  for (int ks = 0; ks < 2; ++ks) {
    FragA a;
#pragma unroll
    for (int vv = 0; vv < 8; ++vv)
      a.u[vv] = Qu[16 * w + li][16 * ks + (vv & 3) + 4 * hi + 8 * (vv >> 2)];
#pragma unroll
    for (int nt = 0; nt < 8; ++nt) {
      FragA bb;
#pragma unroll
      for (int vv = 0; vv < 8; ++vv)
        bb.u[vv] = Ku[16 * nt + li][16 * ks + (vv & 3) + 4 * hi + 8 * (vv >> 2)];
      acc[nt] = __builtin_amdgcn_wmma_f32_16x16x32_f16(false, a.v, false, bb.v, (short)0, acc[nt], false, false);
    }
  }

  // masks + logsumexp per row (rows v+8hi across 16 lanes, 8 N-tiles)
  float logit_store[8];
#pragma unroll
  for (int vv = 0; vv < 8; ++vv) {
    int m  = 16 * w + vv + 8 * hi;
    int qi = tq[m] & (SS - 1);
    float rowmax = -1e30f;
#pragma unroll
    for (int nt = 0; nt < 8; ++nt) {
      int ki = tk[16 * nt + li] & (SS - 1);
      float x = acc[nt][vv];
      if (qi < ki)  x -= 1e9f;   // decoder causal mask on original positions
      if (qi == ki) x -= 1e5f;   // self-attention penalty
      acc[nt][vv] = x;
      rowmax = fmaxf(rowmax, x);
    }
#pragma unroll
    for (int mk = 1; mk < 16; mk <<= 1)
      rowmax = fmaxf(rowmax, __shfl_xor(rowmax, mk, 32));
    float rsum = 0.f;
#pragma unroll
    for (int nt = 0; nt < 8; ++nt) rsum += __expf(acc[nt][vv] - rowmax);
#pragma unroll
    for (int mk = 1; mk < 16; mk <<= 1)
      rsum += __shfl_xor(rsum, mk, 32);
    float logit = rowmax + __logf(rsum);
    logit_store[vv] = logit;
#pragma unroll
    for (int nt = 0; nt < 8; ++nt)
      Ph[m][16 * nt + li] = (_Float16)__expf(acc[nt][vv] - logit);
  }
  if (li == 0) {
#pragma unroll
    for (int vv = 0; vv < 8; ++vv) {
      int m = 16 * w + vv + 8 * hi;
      int t = tq[m];
      int r = t >> 12, pos = t & (SS - 1);
      logitsg[((size_t)bh * NHASH + r) * SS + pos] = logit_store[vv];
    }
  }
  __syncthreads();

  // out = probs(64x128) x V(128x64), K-dim = 128 (4 wmma k-steps)
  const unsigned (*Pu)[64] = (const unsigned (*)[64])Ph;
  const unsigned (*Vu)[64] = (const unsigned (*)[64])Vh;
  v8f oacc[4];
#pragma unroll
  for (int nt = 0; nt < 4; ++nt) oacc[nt] = (v8f){0,0,0,0,0,0,0,0};
#pragma unroll
  for (int ks = 0; ks < 4; ++ks) {
    FragA a;
#pragma unroll
    for (int vv = 0; vv < 8; ++vv)
      a.u[vv] = Pu[16 * w + li][16 * ks + (vv & 3) + 4 * hi + 8 * (vv >> 2)];
#pragma unroll
    for (int nt = 0; nt < 4; ++nt) {
      FragA bb;
#pragma unroll
      for (int vv = 0; vv < 8; ++vv)
        bb.u[vv] = Vu[16 * nt + li][16 * ks + (vv & 3) + 4 * hi + 8 * (vv >> 2)];
      oacc[nt] = __builtin_amdgcn_wmma_f32_16x16x32_f16(false, a.v, false, bb.v, (short)0, oacc[nt], false, false);
    }
  }

  // scatter out rows to (b,h,hash,pos,d) — fuses the undo-sort gather
#pragma unroll
  for (int vv = 0; vv < 8; ++vv) {
    int m = 16 * w + vv + 8 * hi;
    int t = tq[m];
    int r = t >> 12, pos = t & (SS - 1);
    size_t base = (((size_t)bh * NHASH + r) * SS + pos) * DD;
#pragma unroll
    for (int nt = 0; nt < 4; ++nt)
      outh[base + 16 * nt + li] = oacc[nt][vv];
  }
}

// ---------------------------------------------------------------------------
// K5: combine hash rounds with softmax over logits; write (B,S,H*D)
// ---------------------------------------------------------------------------
__global__ __launch_bounds__(256)
void k_combine(const float* __restrict__ outh, const float* __restrict__ logitsg,
               float* __restrict__ out)
{
  int gid = blockIdx.x * 256 + threadIdx.x;     // flat over B*S*H*D = 2^23
  int d = gid & 63;
  int h = (gid >> 6) & 15;
  int s = (gid >> 10) & (SS - 1);
  int b = gid >> 22;
  int bh = b * HH + h;

  float lg[NHASH];
  float mx = -1e30f;
#pragma unroll
  for (int r = 0; r < NHASH; ++r) {
    lg[r] = logitsg[((size_t)bh * NHASH + r) * SS + s];
    mx = fmaxf(mx, lg[r]);
  }
  float sum = 0.f, o = 0.f;
#pragma unroll
  for (int r = 0; r < NHASH; ++r) {
    float p = __expf(lg[r] - mx);
    sum += p;
    o = fmaf(p, outh[(((size_t)bh * NHASH + r) * SS + s) * DD + d], o);
  }
  out[gid] = o / sum;
}

// ---------------------------------------------------------------------------
extern "C" void kernel_launch(void* const* d_in, const int* in_sizes, int n_in,
                              void* d_out, int out_size, void* d_ws, size_t ws_size,
                              hipStream_t stream)
{
  (void)in_sizes; (void)n_in; (void)out_size; (void)ws_size;
  const float* hid = (const float*)d_in[0];
  const float* wqk = (const float*)d_in[1];
  const float* wv  = (const float*)d_in[2];
  const float* rotations = (const float*)d_in[3];
  float* out = (float*)d_out;

  char* ws = (char*)d_ws;
  size_t off = 0;
  auto alloc = [&](size_t bytes) -> void* {
    void* p = ws + off;
    off += (bytes + 255) & ~(size_t)255;
    return p;
  };
  float* qk      = (float*)alloc((size_t)BH * SS * DD * 4);          // 33.5 MB
  float* vv      = (float*)alloc((size_t)BH * SS * DD * 4);          // 33.5 MB
  int*   buckets = (int*)  alloc((size_t)BH * LL * 4);               //  2 MB
  int*   hist    = (int*)  alloc((size_t)BH * TBUCK * 4);            // 32 KB
  int*   sorted  = (int*)  alloc((size_t)BH * LL * 4);               //  2 MB
  float* outh    = (float*)alloc((size_t)BH * NHASH * SS * DD * 4);  // 134 MB
  float* logitsg = (float*)alloc((size_t)BH * NHASH * SS * 4);       //  2 MB

  k_zero<<<dim3((BH * TBUCK + 255) / 256), 256, 0, stream>>>(hist, BH * TBUCK);
  k_proj<<<dim3(128, 16), 256, 0, stream>>>(hid, wqk, wv, qk, vv);
  k_buckets<<<dim3((BH * SS) / 256), 256, 0, stream>>>(qk, rotations, buckets, hist);
  k_sort<<<dim3(BH), 256, 0, stream>>>(buckets, hist, sorted);
  k_attn<<<dim3(NCHUNK, BH), 128, 0, stream>>>(qk, vv, sorted, outh, logitsg);
  k_combine<<<dim3((BB * SS * HH * DD) / 256), 256, 0, stream>>>(outh, logitsg, out);
}
